// HrmAttention_62826781606440
// MI455X (gfx1250) — compile-verified, matrix operand order checked by reading, not run on hardware
//
#include <hip/hip_runtime.h>
#include <hip/hip_bf16.h>

// Problem constants (from reference): B=4, S=2048, D=2048, H=16, HD=128
#define BB 4
#define SS 2048
#define DD 2048
#define HH 16
#define HD 128

typedef __attribute__((ext_vector_type(16))) _Float16 v16h;
typedef __attribute__((ext_vector_type(8)))  _Float16 v8h;
typedef __attribute__((ext_vector_type(8)))  float    v8f;

// ---------------------------------------------------------------------------
// CDNA5 async copy: per-lane 16B global -> LDS, tracked by ASYNCcnt.
// Generic LDS pointer low 32 bits are the LDS byte offset (flat aperture
// truncation), so a u32 cast gives the VDST LDS address.
// ---------------------------------------------------------------------------
__device__ __forceinline__ void async_cp_b128(const void* g, void* l) {
    asm volatile("global_load_async_to_lds_b128 %0, %1, off"
                 :: "v"((unsigned)(unsigned long long)l),
                    "v"((unsigned long long)g)
                 : "memory");
}
__device__ __forceinline__ void wait_async0() {
    asm volatile("s_wait_asynccnt 0" ::: "memory");
}

// ---------------------------------------------------------------------------
// WMMA f16 16x16x32 fragment loader (works for global or LDS pointers).
// A-frag (lane = M row) and B-frag (lane = N column) share the per-lane
// pattern: lane<16 holds K in {0..7, 16..23}, lane>=16 holds {8..15, 24..31};
// i.e. two contiguous 16B loads at +ko and +16+ko halves, ko = (lane>>4)*8.
// ---------------------------------------------------------------------------
__device__ __forceinline__ v16h load_frag16(const _Float16* __restrict__ base, int ld) {
    const int lane = threadIdx.x & 31;
    const int r  = lane & 15;
    const int ko = (lane >> 4) * 8;
    const _Float16* p = base + (long)r * ld + ko;
    v8h lo = *(const v8h*)(p);
    v8h hi = *(const v8h*)(p + 16);
    v16h f;
#pragma unroll
    for (int i = 0; i < 8; ++i) { f[i] = lo[i]; f[i + 8] = hi[i]; }
    return f;
}

__device__ __forceinline__ v8f wmma_f16(v16h a, v16h b, v8f c) {
    return __builtin_amdgcn_wmma_f32_16x16x32_f16(false, a, false, b, (short)0, c, false, false);
}

// ---------------------------------------------------------------------------
// fp32 -> fp16 conversion (vectorized x4)
// ---------------------------------------------------------------------------
__global__ void cvt_f32_f16_x4(const float* __restrict__ src,
                               _Float16* __restrict__ dst, long n4) {
    long i = (long)blockIdx.x * blockDim.x + threadIdx.x;
    if (i >= n4) return;
    const float4 v = ((const float4*)src)[i];
    __attribute__((ext_vector_type(4))) _Float16 h;
    h[0] = (_Float16)v.x; h[1] = (_Float16)v.y;
    h[2] = (_Float16)v.z; h[3] = (_Float16)v.w;
    *(__attribute__((ext_vector_type(4))) _Float16*)(dst + i * 4) = h;
}

// ---------------------------------------------------------------------------
// GEMM: C[m,n] = sum_k A[m,k] * Bw[n,k]  (A: MxK row-major, Bw: NxK row-major)
// Block: 256 thr = 8 waves; block tile 128x128, wave tile 64x32, K-step 32.
// A/B K-chunks are staged in LDS via async b128 copies, double-buffered.
// Row pad +8 halves (stride 40 halves = 20 DW) -> conflict-free b128 reads.
// ---------------------------------------------------------------------------
template <typename CT>
__global__ __launch_bounds__(256) void wmma_gemm_nt(
        const _Float16* __restrict__ A, const _Float16* __restrict__ Bw,
        CT* __restrict__ C, int M, int N, int K) {
    __shared__ __align__(16) _Float16 As[2][128][40];
    __shared__ __align__(16) _Float16 Bs[2][128][40];

    const int t    = threadIdx.x;
    const int wave = t >> 5;
    const int lane = t & 31;
    const int mb = blockIdx.y * 128, nb = blockIdx.x * 128;
    const int wm = (wave >> 2) * 64;    // wave row (2 rows of waves)
    const int wn = (wave & 3) * 32;     // wave col (4 cols of waves)

    v8f acc[4][2] = {};

    // each thread copies 2 A pieces + 2 B pieces (128 rows * 4 x b128 each)
    auto copy_chunk = [&](int k0, int buf) {
#pragma unroll
        for (int j = 0; j < 2; ++j) {
            const int idx = t + j * 256;
            const int row = idx >> 2;
            const int pc  = (idx & 3) * 8;
            async_cp_b128(A  + (long)(mb + row) * K + k0 + pc, &As[buf][row][pc]);
            async_cp_b128(Bw + (long)(nb + row) * K + k0 + pc, &Bs[buf][row][pc]);
        }
    };

    copy_chunk(0, 0);
    const int nk = K >> 5;
    for (int it = 0; it < nk; ++it) {
        const int cb = it & 1;
        wait_async0();
        __syncthreads();                       // chunk cb resident; prev reads done
        if (it + 1 < nk) copy_chunk((it + 1) * 32, cb ^ 1);

        v16h bf[2];
#pragma unroll
        for (int nt = 0; nt < 2; ++nt)
            bf[nt] = load_frag16(&Bs[cb][wn + nt * 16][0], 40);
#pragma unroll
        for (int mt = 0; mt < 4; ++mt) {
            v16h af = load_frag16(&As[cb][wm + mt * 16][0], 40);
#pragma unroll
            for (int nt = 0; nt < 2; ++nt)
                acc[mt][nt] = wmma_f16(af, bf[nt], acc[mt][nt]);
        }
    }

    // C/D layout: lane holds N = lane&15, VGPR r -> M = r + 8*(lane>>4)
    const int cn = lane & 15;
    const int rb = (lane >> 4) * 8;
#pragma unroll
    for (int mt = 0; mt < 4; ++mt)
#pragma unroll
        for (int nt = 0; nt < 2; ++nt)
#pragma unroll
            for (int r = 0; r < 8; ++r) {
                long idx = (long)(mb + wm + mt * 16 + rb + r) * N + (nb + wn + nt * 16 + cn);
                C[idx] = (CT)acc[mt][nt][r];
            }
}

// ---------------------------------------------------------------------------
// RoPE + split + head-major repack.
//   qkv (f16, [B*S, 3*D]) -> Qh [b,h,s,hd] (scaled by 1/sqrt(HD)),
//                            Kh [b,h,s,hd], Vt [b,h,hd,s]
// ---------------------------------------------------------------------------
__global__ void rope_split(const _Float16* __restrict__ qkv,
                           const float* __restrict__ cosb,
                           const float* __restrict__ sinb,
                           _Float16* __restrict__ Qh,
                           _Float16* __restrict__ Kh,
                           _Float16* __restrict__ Vt) {
    long i = (long)blockIdx.x * blockDim.x + threadIdx.x;
    const long total = (long)BB * SS * HH * HD;
    if (i >= total) return;
    const int hd = (int)(i & (HD - 1));
    const int h  = (int)((i >> 7) & (HH - 1));
    const int s  = (int)((i >> 11) & (SS - 1));
    const int b  = (int)(i >> 22);

    const long row = ((long)b * SS + s) * (3L * DD);
    const int  col = h * HD + hd;
    const float c  = cosb[s * HD + hd];
    const float sn = sinb[s * HD + hd];

    const int  phd = (hd < 64) ? hd + 64 : hd - 64;
    const float sg = (hd < 64) ? -1.0f : 1.0f;

    float q  = (float)qkv[row + col];
    float k  = (float)qkv[row + DD + col];
    float v  = (float)qkv[row + 2 * DD + col];
    float qp = (float)qkv[row + h * HD + phd];
    float kp = (float)qkv[row + DD + h * HD + phd];

    const float scale = 0.08838834764831845f;  // 1/sqrt(128)
    float qr = (q * c + sg * qp * sn) * scale;
    float kr = (k * c + sg * kp * sn);

    const long bh = (long)b * HH + h;
    Qh[(bh * SS + s) * HD + hd] = (_Float16)qr;
    Kh[(bh * SS + s) * HD + hd] = (_Float16)kr;
    Vt[(bh * HD + hd) * SS + s] = (_Float16)v;
}

// ---------------------------------------------------------------------------
// Flash attention, transposed-score formulation.
//   S^T = K x Q^T; softmax over kcols = in-lane reduce + shfl_xor(16);
//   P^T B-frag built in-lane from the two S^T C-tiles; attn^T = V^T x P^T.
// Block: 256 thr = 8 waves; blockIdx.y = b*H+h, blockIdx.x = 256-wide q block;
// each wave owns 32 q columns. K/V 32-kcol chunks are shared by all 8 waves ->
// staged once per block in LDS via async copies (double-buffered).
// ---------------------------------------------------------------------------
__global__ __launch_bounds__(256) void attn_kernel(
        const _Float16* __restrict__ Qh, const _Float16* __restrict__ Kh,
        const _Float16* __restrict__ Vt, _Float16* __restrict__ attnH) {
    __shared__ __align__(16) _Float16 Ks[2][32][136];   // [kcol][hd] (+8 pad)
    __shared__ __align__(16) _Float16 Vs[2][128][40];   // [hd][kcol] (+8 pad)

    const int t    = threadIdx.x;
    const int wave = t >> 5;
    const int lane = t & 31;
    const int bh = blockIdx.y;
    const int b = bh >> 4, h = bh & 15;
    const int q0 = blockIdx.x * 256 + wave * 32;

    const _Float16* Qb = Qh + (long)bh * SS * HD;   // [s][hd]
    const _Float16* Kb = Kh + (long)bh * SS * HD;   // [s][hd]
    const _Float16* Vb = Vt + (long)bh * HD * SS;   // [hd][s]

    // K chunk: 32 rows * 16 x b128; V chunk: 128 rows * 4 x b128 (2+2/thread)
    auto copy_chunk = [&](int kc, int buf) {
#pragma unroll
        for (int j = 0; j < 2; ++j) {
            const int idx = t + j * 256;
            const int krow = idx >> 4, kpc = (idx & 15) * 8;
            async_cp_b128(Kb + (long)(kc + krow) * HD + kpc, &Ks[buf][krow][kpc]);
            const int vrow = idx >> 2, vpc = (idx & 3) * 8;
            async_cp_b128(Vb + (long)vrow * SS + kc + vpc, &Vs[buf][vrow][vpc]);
        }
    };

    // Preload Q B-fragments: 2 q-tiles x 4 K-steps (HD = 128 = 4*32)
    v16h qf[2][4];
#pragma unroll
    for (int nt = 0; nt < 2; ++nt)
#pragma unroll
        for (int ks = 0; ks < 4; ++ks)
            qf[nt][ks] = load_frag16(Qb + (long)(q0 + nt * 16) * HD + ks * 32, HD);

    v8f oacc[8][2] = {};                     // attn^T acc: 8 hd-tiles x 2 q-tiles
    float mrun[2] = {-3.0e38f, -3.0e38f};
    float lrun[2] = {0.0f, 0.0f};

    copy_chunk(0, 0);
    for (int it = 0; it < SS / 32; ++it) {
        const int cb = it & 1;
        wait_async0();
        __syncthreads();
        if (it + 1 < SS / 32) copy_chunk((it + 1) * 32, cb ^ 1);

        // ---- scores chunk S^T[32 kcols x 32 q] from LDS ----
        v8f sacc[2][2] = {};
#pragma unroll
        for (int ks = 0; ks < 4; ++ks) {
            v16h ka0 = load_frag16(&Ks[cb][0][ks * 32], 136);
            v16h ka1 = load_frag16(&Ks[cb][16][ks * 32], 136);
#pragma unroll
            for (int nt = 0; nt < 2; ++nt) {
                sacc[0][nt] = wmma_f16(ka0, qf[nt][ks], sacc[0][nt]);
                sacc[1][nt] = wmma_f16(ka1, qf[nt][ks], sacc[1][nt]);
            }
        }

        // ---- online softmax + build P^T B-fragments (pure in-lane) ----
        v16h pfrag[2];
#pragma unroll
        for (int nt = 0; nt < 2; ++nt) {
            float cmax = -3.0e38f;
#pragma unroll
            for (int r = 0; r < 8; ++r) {
                cmax = fmaxf(cmax, sacc[0][nt][r]);
                cmax = fmaxf(cmax, sacc[1][nt][r]);
            }
            cmax = fmaxf(cmax, __shfl_xor(cmax, 16, 32));   // combine lane pair
            const float mnew = fmaxf(mrun[nt], cmax);
            const float f = __expf(mrun[nt] - mnew);
            mrun[nt] = mnew;
            lrun[nt] *= f;
            float psum = 0.0f;
            v16h pf;
#pragma unroll
            for (int r = 0; r < 8; ++r) {
                float e0 = __expf(sacc[0][nt][r] - mnew);
                float e1 = __expf(sacc[1][nt][r] - mnew);
                psum += e0 + e1;
                pf[r]     = (_Float16)e0;   // K = r  + 8*(lane>>4)
                pf[r + 8] = (_Float16)e1;   // K = 16 + r + 8*(lane>>4)
            }
            lrun[nt] += psum;
            pfrag[nt] = pf;
#pragma unroll
            for (int mt = 0; mt < 8; ++mt)
#pragma unroll
                for (int r = 0; r < 8; ++r)
                    oacc[mt][nt][r] *= f;
        }

        // ---- attn^T += V^T(128 x 32) x P^T(32 x 32), V from LDS ----
#pragma unroll
        for (int mt = 0; mt < 8; ++mt) {
            v16h vf = load_frag16(&Vs[cb][mt * 16][0], 40);
#pragma unroll
            for (int nt = 0; nt < 2; ++nt)
                oacc[mt][nt] = wmma_f16(vf, pfrag[nt], oacc[mt][nt]);
        }
    }

    // ---- finalize + store attn as f16 [m, d]; 8 contiguous hd per b128 ----
    const int cn = lane & 15;
    const int rb = (lane >> 4) * 8;
    float rl[2];
#pragma unroll
    for (int nt = 0; nt < 2; ++nt) {
        float l = lrun[nt] + __shfl_xor(lrun[nt], 16, 32);
        rl[nt] = 1.0f / l;
    }
#pragma unroll
    for (int mt = 0; mt < 8; ++mt)
#pragma unroll
        for (int nt = 0; nt < 2; ++nt) {
            v8h o;
#pragma unroll
            for (int r = 0; r < 8; ++r)
                o[r] = (_Float16)(oacc[mt][nt][r] * rl[nt]);
            const long row = (long)b * SS + (q0 + nt * 16 + cn);
            const int  col = h * HD + mt * 16 + rb;
            *(v8h*)(attnH + row * DD + col) = o;
        }
}

// ---------------------------------------------------------------------------
extern "C" void kernel_launch(void* const* d_in, const int* in_sizes, int n_in,
                              void* d_out, int out_size, void* d_ws, size_t ws_size,
                              hipStream_t stream) {
    const float* hidden = (const float*)d_in[0];   // [B,S,D]
    const float* cosb   = (const float*)d_in[1];   // [S,HD]
    const float* sinb   = (const float*)d_in[2];   // [S,HD]
    const float* w_qkv  = (const float*)d_in[3];   // [3D,D]
    const float* w_o    = (const float*)d_in[4];   // [D,D]
    float* out = (float*)d_out;                    // [B,S,D]

    const long M  = (long)BB * SS;        // 8192
    const long K  = DD;                   // 2048
    const long N1 = 3L * DD;              // 6144

    char* ws = (char*)d_ws;
    _Float16* Xh    = (_Float16*)ws;  ws += M * K * 2;            // 33.5 MB
    _Float16* Wqkv  = (_Float16*)ws;  ws += N1 * K * 2;           // 25.2 MB
    _Float16* Wo    = (_Float16*)ws;  ws += (long)DD * DD * 2;    //  8.4 MB
    _Float16* QKV   = (_Float16*)ws;  ws += M * N1 * 2;           // 100.7 MB
    _Float16* Qh    = (_Float16*)ws;  ws += M * DD * 2;           // 33.5 MB
    _Float16* Kh    = (_Float16*)ws;  ws += M * DD * 2;           // 33.5 MB
    _Float16* Vt    = (_Float16*)ws;  ws += M * DD * 2;           // 33.5 MB
    _Float16* attnH = Xh;  // Xh is dead after GEMM1 -> reuse for attn output

    // 1) f32 -> f16 conversions
    {
        long n4;
        n4 = (M * K) / 4;
        cvt_f32_f16_x4<<<dim3((unsigned)((n4 + 255) / 256)), 256, 0, stream>>>(hidden, Xh, n4);
        n4 = (N1 * K) / 4;
        cvt_f32_f16_x4<<<dim3((unsigned)((n4 + 255) / 256)), 256, 0, stream>>>(w_qkv, Wqkv, n4);
        n4 = ((long)DD * DD) / 4;
        cvt_f32_f16_x4<<<dim3((unsigned)((n4 + 255) / 256)), 256, 0, stream>>>(w_o, Wo, n4);
    }

    // 2) QKV projection: [8192,2048] x [6144,2048]^T -> f16 [8192,6144]
    wmma_gemm_nt<_Float16><<<dim3((unsigned)(N1 / 128), (unsigned)(M / 128)),
                             256, 0, stream>>>(Xh, Wqkv, QKV, (int)M, (int)N1, (int)K);

    // 3) RoPE + split + repack (Q pre-scaled by 1/sqrt(HD), V transposed)
    {
        long total = (long)BB * SS * HH * HD;   // 16.8M threads
        rope_split<<<dim3((unsigned)((total + 255) / 256)), 256, 0, stream>>>(
            QKV, cosb, sinb, Qh, Kh, Vt);
    }

    // 4) Flash attention: 64 (b,h) x 8 q-blocks
    attn_kernel<<<dim3(SS / 256, BB * HH), 256, 0, stream>>>(Qh, Kh, Vt, attnH);

    // 5) Output projection: [8192,2048] x [2048,2048]^T -> f32 out
    wmma_gemm_nt<float><<<dim3(DD / 128, (unsigned)(M / 128)),
                          256, 0, stream>>>(attnH, Wo, out, (int)M, (int)DD, (int)K);
}
